// GINENet_33208687133064
// MI455X (gfx1250) — compile-verified, matrix operand order checked by reading, not run on hardware
//
#include <hip/hip_runtime.h>
#include <hip/hip_bf16.h>
#include <stdint.h>

typedef _Float16 v16h __attribute__((ext_vector_type(16)));
typedef _Float16 v8h  __attribute__((ext_vector_type(8)));
typedef float    v8f  __attribute__((ext_vector_type(8)));

#define F_DIM 128

// ---------------------------------------------------------------------------
// Zero a float buffer (graph-capture-safe replacement for hipMemsetAsync)
// ---------------------------------------------------------------------------
__global__ void zero_f4(float4* __restrict__ p, int n4) {
    int i = blockIdx.x * blockDim.x + threadIdx.x;
    if (i < n4) p[i] = make_float4(0.f, 0.f, 0.f, 0.f);
}

// ---------------------------------------------------------------------------
// Pre-swizzle the four 128x128 fp32 weight matrices into f16 WMMA B-fragment
// layout: frag[(kt*8+nt)*32 + lane] = 16 halves (8 dwords), where per the
// CDNA5 16-bit B layout lane&15 selects N, lane>>4 selects the K half-window.
// Dword d<4 holds K = kb+2d, kb+2d+1 ; d>=4 holds K = kb+16+2(d-4), +1.
// ---------------------------------------------------------------------------
__global__ void wconvert(const float* __restrict__ W11, const float* __restrict__ W12,
                         const float* __restrict__ W21, const float* __restrict__ W22,
                         _Float16* __restrict__ frags) {
    int gid = blockIdx.x * blockDim.x + threadIdx.x;     // 4 * 8192 dwords total
    if (gid >= 4 * 8192) return;
    int m    = gid >> 13;          // matrix id
    int r    = gid & 8191;         // dword index inside this matrix's frag array
    int d    = r & 7;
    int lane = (r >> 3) & 31;
    int ntk  = r >> 8;             // kt*8 + nt
    int kt   = ntk >> 3;
    int nt   = ntk & 7;
    int nn   = nt * 16 + (lane & 15);
    int kb   = kt * 32 + ((lane >> 4) << 3);
    int k0   = kb + ((d < 4) ? (d << 1) : (16 + ((d - 4) << 1)));
    const float* W = (m == 0) ? W11 : (m == 1) ? W12 : (m == 2) ? W21 : W22;
    _Float16* o = frags + (size_t)m * 16384 + (size_t)r * 2;
    o[0] = (_Float16)W[(size_t)k0 * F_DIM + nn];
    o[1] = (_Float16)W[(size_t)(k0 + 1) * F_DIM + nn];
}

// ---------------------------------------------------------------------------
// Edge phase: msg = relu(x[src] + edge_attr*We + be); atomicAdd into aggr[dst]
// One wave per edge (32 lanes x float4 = the full 128-feature row, coalesced).
// ---------------------------------------------------------------------------
__global__ void edge_scatter(const float* __restrict__ xin, const int* __restrict__ ei,
                             const float* __restrict__ ea, const float* __restrict__ We,
                             const float* __restrict__ be, float* aggr, int E_) {
    long long gid = (long long)blockIdx.x * blockDim.x + threadIdx.x;
    if (gid >= (long long)E_ * 32) return;
    int e = (int)(gid >> 5);
    int c = ((int)gid & 31) << 2;           // feature chunk of 4
    int src = ei[e];
    int dst = ei[E_ + e];
    float a = ea[e];
    const float* xr = xin + (size_t)src * F_DIM + c;
    float4 xv = *(const float4*)xr;
    float4 wv = *(const float4*)(We + c);
    float4 bv = *(const float4*)(be + c);
    float m0 = fmaxf(xv.x + a * wv.x + bv.x, 0.f);
    float m1 = fmaxf(xv.y + a * wv.y + bv.y, 0.f);
    float m2 = fmaxf(xv.z + a * wv.z + bv.z, 0.f);
    float m3 = fmaxf(xv.w + a * wv.w + bv.w, 0.f);
    float* o = aggr + (size_t)dst * F_DIM + c;
    atomicAdd(o + 0, m0);
    atomicAdd(o + 1, m1);
    atomicAdd(o + 2, m2);
    atomicAdd(o + 3, m3);
}

// ---------------------------------------------------------------------------
// Fused node MLP: out = relu( relu(((1+eps)*x + aggr) @ W1 + b1) @ W2 + b2 )
// Each wave owns 16 rows. Activations staged in a private LDS tile (f16,
// row pitch 136 halves -> conflict-free ds_load_b128 A fragments), weights
// read as pre-swizzled B fragments (two global_load_b128, L0-resident).
// 64 v_wmma_f32_16x16x32_f16 per wave. No cross-wave synchronization.
// ---------------------------------------------------------------------------
__global__ void __launch_bounds__(256)
mlp_wmma(const float* __restrict__ xin, const float* aggr, const float* __restrict__ eps,
         const _Float16* __restrict__ fw1, const float* __restrict__ b1,
         const _Float16* __restrict__ fw2, const float* __restrict__ b2,
         float* out, int nrows) {
    __shared__ __align__(16) _Float16 stage[8][16 * 136];

    const int wv   = threadIdx.x >> 5;
    const int lane = threadIdx.x & 31;
    const int row0 = blockIdx.x * 128 + wv * 16;
    if (row0 >= nrows) return;

    const float ep = 1.0f + eps[0];
    _Float16* st = &stage[wv][0];

    // ---- stage h = (1+eps)*x + aggr as f16 (b64 LDS stores, conflict-free)
    #pragma unroll 4
    for (int r = 0; r < 16; ++r) {
        size_t off = (size_t)(row0 + r) * F_DIM + (lane << 2);
        float4 xv = *(const float4*)(xin + off);
        float4 av = *(const float4*)(aggr + off);
        union { _Float16 h[4]; unsigned long long u; } p;
        p.h[0] = (_Float16)(ep * xv.x + av.x);
        p.h[1] = (_Float16)(ep * xv.y + av.y);
        p.h[2] = (_Float16)(ep * xv.z + av.z);
        p.h[3] = (_Float16)(ep * xv.w + av.w);
        *(unsigned long long*)(st + r * 136 + (lane << 2)) = p.u;
    }

    const int m_  = lane & 15;
    const int hs8 = (lane >> 4) << 3;

    v8f acc[8];
    const v8f vz = {0.f, 0.f, 0.f, 0.f, 0.f, 0.f, 0.f, 0.f};

    // ================= layer 1 =================
    #pragma unroll
    for (int nt = 0; nt < 8; ++nt) acc[nt] = vz;
    #pragma unroll
    for (int kt = 0; kt < 4; ++kt) {
        int kb = kt * 32 + hs8;
        v8h alo = *(const v8h*)(st + m_ * 136 + kb);
        v8h ahi = *(const v8h*)(st + m_ * 136 + kb + 16);
        v16h av = __builtin_shufflevector(alo, ahi, 0,1,2,3,4,5,6,7,8,9,10,11,12,13,14,15);
        #pragma unroll
        for (int nt = 0; nt < 8; ++nt) {
            const _Float16* q = fw1 + ((size_t)((kt << 3) | nt) * 32 + lane) * 16;
            v8h blo = *(const v8h*)q;
            v8h bhi = *(const v8h*)(q + 8);
            v16h bv = __builtin_shufflevector(blo, bhi, 0,1,2,3,4,5,6,7,8,9,10,11,12,13,14,15);
            acc[nt] = __builtin_amdgcn_wmma_f32_16x16x32_f16(
                false, av, false, bv, (short)0, acc[nt], false, false);
        }
    }

    // ---- bias + relu, restage as f16 for layer 2
    #pragma unroll
    for (int nt = 0; nt < 8; ++nt) {
        int col = nt * 16 + m_;
        float bs = b1[col];
        #pragma unroll
        for (int r = 0; r < 8; ++r) {
            float v = fmaxf(acc[nt][r] + bs, 0.f);
            st[(r + hs8) * 136 + col] = (_Float16)v;
        }
    }

    // ================= layer 2 =================
    #pragma unroll
    for (int nt = 0; nt < 8; ++nt) acc[nt] = vz;
    #pragma unroll
    for (int kt = 0; kt < 4; ++kt) {
        int kb = kt * 32 + hs8;
        v8h alo = *(const v8h*)(st + m_ * 136 + kb);
        v8h ahi = *(const v8h*)(st + m_ * 136 + kb + 16);
        v16h av = __builtin_shufflevector(alo, ahi, 0,1,2,3,4,5,6,7,8,9,10,11,12,13,14,15);
        #pragma unroll
        for (int nt = 0; nt < 8; ++nt) {
            const _Float16* q = fw2 + ((size_t)((kt << 3) | nt) * 32 + lane) * 16;
            v8h blo = *(const v8h*)q;
            v8h bhi = *(const v8h*)(q + 8);
            v16h bv = __builtin_shufflevector(blo, bhi, 0,1,2,3,4,5,6,7,8,9,10,11,12,13,14,15);
            acc[nt] = __builtin_amdgcn_wmma_f32_16x16x32_f16(
                false, av, false, bv, (short)0, acc[nt], false, false);
        }
    }

    // ---- bias + relu epilogue, write fp32
    #pragma unroll
    for (int nt = 0; nt < 8; ++nt) {
        int col = nt * 16 + m_;
        float bs = b2[col];
        #pragma unroll
        for (int r = 0; r < 8; ++r) {
            out[(size_t)(row0 + r + hs8) * F_DIM + col] = fmaxf(acc[nt][r] + bs, 0.f);
        }
    }
}

// ---------------------------------------------------------------------------
// Classifier + log_softmax: logits = h @ Wl + bl (128 -> 40), per-row softmax.
// Wl kept in LDS (broadcast reads); one thread per node.
// ---------------------------------------------------------------------------
__global__ void __launch_bounds__(256)
classifier(const float* __restrict__ h, const float* __restrict__ Wl,
           const float* __restrict__ bl, float* __restrict__ out, int n) {
    __shared__ float w[F_DIM * 40];
    __shared__ float bb[40];
    for (int i = threadIdx.x; i < F_DIM * 40; i += blockDim.x) w[i] = Wl[i];
    if (threadIdx.x < 40) bb[threadIdx.x] = bl[threadIdx.x];
    __syncthreads();

    int node = blockIdx.x * blockDim.x + threadIdx.x;
    if (node >= n) return;

    float acc[40];
    #pragma unroll
    for (int c = 0; c < 40; ++c) acc[c] = bb[c];

    const float* hr = h + (size_t)node * F_DIM;
    for (int k = 0; k < F_DIM; ++k) {
        float hv = hr[k];
        #pragma unroll
        for (int c = 0; c < 40; ++c) acc[c] = fmaf(hv, w[k * 40 + c], acc[c]);
    }

    float mx = acc[0];
    #pragma unroll
    for (int c = 1; c < 40; ++c) mx = fmaxf(mx, acc[c]);
    float s = 0.f;
    #pragma unroll
    for (int c = 0; c < 40; ++c) s += __expf(acc[c] - mx);
    float ls = __logf(s);
    float* o = out + (size_t)node * 40;
    #pragma unroll
    for (int c = 0; c < 40; ++c) o[c] = acc[c] - mx - ls;
}

// ---------------------------------------------------------------------------
extern "C" void kernel_launch(void* const* d_in, const int* in_sizes, int n_in,
                              void* d_out, int out_size, void* d_ws, size_t ws_size,
                              hipStream_t stream) {
    const float* x    = (const float*)d_in[0];
    const int*   ei   = (const int*)d_in[1];
    const float* ea   = (const float*)d_in[2];
    const float* eps1 = (const float*)d_in[3];
    const float* We1  = (const float*)d_in[4];
    const float* be1  = (const float*)d_in[5];
    const float* W11  = (const float*)d_in[6];
    const float* b11  = (const float*)d_in[7];
    const float* W12  = (const float*)d_in[8];
    const float* b12  = (const float*)d_in[9];
    const float* eps2 = (const float*)d_in[10];
    const float* We2  = (const float*)d_in[11];
    const float* be2  = (const float*)d_in[12];
    const float* W21  = (const float*)d_in[13];
    const float* b21  = (const float*)d_in[14];
    const float* W22  = (const float*)d_in[15];
    const float* b22  = (const float*)d_in[16];
    const float* Wl   = (const float*)d_in[17];
    const float* bl   = (const float*)d_in[18];

    const int N_ = in_sizes[0] / F_DIM;
    const int E_ = in_sizes[2];
    const size_t NB = (size_t)N_ * F_DIM * sizeof(float);

    char* ws = (char*)d_ws;
    float*    aggr  = (float*)ws;             // N x 128 f32
    float*    h1    = (float*)(ws + NB);      // N x 128 f32
    _Float16* frags = (_Float16*)(ws + 2 * NB); // 4 x 16384 halves (128 KB)
    _Float16* f11 = frags;
    _Float16* f12 = frags + 16384;
    _Float16* f21 = frags + 32768;
    _Float16* f22 = frags + 49152;
    float* h2  = aggr;                        // safe alias: per-wave read-before-write
    float* out = (float*)d_out;

    const int n4 = N_ * F_DIM / 4;
    const int zb = (n4 + 255) / 256;
    const long long ework = (long long)E_ * 32;
    const int eb = (int)((ework + 255) / 256);
    const int mb = (N_ + 127) / 128;
    const int cb = (N_ + 255) / 256;

    wconvert<<<128, 256, 0, stream>>>(W11, W12, W21, W22, frags);

    zero_f4<<<zb, 256, 0, stream>>>((float4*)aggr, n4);
    edge_scatter<<<eb, 256, 0, stream>>>(x, ei, ea, We1, be1, aggr, E_);
    mlp_wmma<<<mb, 256, 0, stream>>>(x, aggr, eps1, f11, b11, f12, b12, h1, N_);

    zero_f4<<<zb, 256, 0, stream>>>((float4*)aggr, n4);
    edge_scatter<<<eb, 256, 0, stream>>>(h1, ei, ea, We2, be2, aggr, E_);
    mlp_wmma<<<mb, 256, 0, stream>>>(h1, aggr, eps2, f21, b21, f22, b22, h2, N_);

    classifier<<<cb, 256, 0, stream>>>(h2, Wl, bl, out, N_);
}
